// GAT_57964878626982
// MI455X (gfx1250) — matrix operand order repaired and checked by reading, not compiled, not run on hardware
//
#include <hip/hip_runtime.h>
#include <hip/hip_bf16.h>
#include <math.h>

// ---------------------------------------------------------------------------
// GAT 2-layer forward for gfx1250 (MI455X).
//   GEMM1 (20000x128 @ 128x512, compute-dense): v_wmma_f32_16x16x32_bf16,
//     bf16x3 splitting for ~fp32 accuracy, 4 N-tiles/wave (12 WMMA / K-chunk).
//   GEMM2 (20000x512 @ 512x2, ~1 FLOP/byte): memory-bound -> wave-per-node
//     dot products with wave32 shfl_xor reduction, fused with attention scores.
//   Edge softmax/aggregation: 3 atomic passes, resident in 192MB L2.
// ---------------------------------------------------------------------------

typedef __bf16 v16bf __attribute__((ext_vector_type(16)));
typedef float  v8f   __attribute__((ext_vector_type(8)));

// ---------------- fill ----------------
__global__ void fill_u32_kernel(unsigned* __restrict__ p, unsigned v, long long n) {
  long long i = (long long)blockIdx.x * blockDim.x + threadIdx.x;
  if (i < n) p[i] = v;
}

// ---------------- GEMM1: C[M,N] = A[M,K] * B[K,N], N%16==0, K%32==0 ---------
// One 16-row block x NT 16-col tiles per wave; K consumed 32/step via WMMA.
// bf16x3: A=Ah+Al, B=Bh+Bl; acc += Ah*Bh + Ah*Bl + Al*Bh  (error ~2^-16 rel).
template <int NT>
__global__ __launch_bounds__(256) void gemm_wmma_bf16x3(
    const float* __restrict__ A, const float* __restrict__ B,
    float* __restrict__ C, int M, int N, int K, int ldA, int ldB, int ldC)
{
  const int lane = threadIdx.x & 31;
  const int waveInBlk = threadIdx.x >> 5;
  const int tilesM = (M + 15) >> 4;
  const int nGroups = (N >> 4) / NT;          // N is a multiple of 16*NT
  long long t = (long long)blockIdx.x * (blockDim.x >> 5) + waveInBlk;
  if (t >= (long long)tilesM * nGroups) return;
  const int tm = (int)(t / nGroups), tg = (int)(t % nGroups);
  const int m0 = tm << 4;
  const int half = lane >> 4;           // K-group selector for A, M-group for C
  const int mr = lane & 15;

  int arow = m0 + mr;                   // A row held by this lane (ISA layout)
  if (arow >= M) arow = M - 1;          // safe clamp; garbage rows masked at store

  v8f acc[NT] = {};

  for (int kc = 0; kc < K; kc += 32) {
    // ---- A fragment (16x32 bf16, ISA layout): lane L holds row M=L%16,
    //      halfwords e=0..7  -> K = kc + 8*half + e
    //      halfwords e=8..15 -> K = kc + 16 + 8*half + (e-8)
    v16bf ah, al;
    {
      const float* ap = A + (long long)arow * ldA + kc + half * 8;
      float4 f0 = *(const float4*)(ap);
      float4 f1 = *(const float4*)(ap + 4);
      float4 f2 = *(const float4*)(ap + 16);
      float4 f3 = *(const float4*)(ap + 20);
      float fa[16] = { f0.x, f0.y, f0.z, f0.w, f1.x, f1.y, f1.z, f1.w,
                       f2.x, f2.y, f2.z, f2.w, f3.x, f3.y, f3.z, f3.w };
#pragma unroll
      for (int e = 0; e < 16; ++e) {
        __bf16 hi = (__bf16)fa[e];
        ah[e] = hi;
        al[e] = (__bf16)(fa[e] - (float)hi);
      }
    }
    // ---- B fragments (32x16 bf16): lane L holds N = n0+L%16, K same as A fragment
    const float* brow = B + (long long)(kc + half * 8) * ldB + mr;
#pragma unroll
    for (int j = 0; j < NT; ++j) {
      const int n0 = (tg * NT + j) << 4;
      v16bf bh, bl;
      const float* bp = brow + n0;
      float fb[16];
#pragma unroll
      for (int e = 0; e < 16; ++e)
        fb[e] = bp[(long long)((e & 7) + ((e >> 3) << 4)) * ldB];
#pragma unroll
      for (int e = 0; e < 16; ++e) {
        __bf16 hi = (__bf16)fb[e];
        bh[e] = hi;
        bl[e] = (__bf16)(fb[e] - (float)hi);
      }
      acc[j] = __builtin_amdgcn_wmma_f32_16x16x32_bf16(false, ah, false, bh, (short)0, acc[j], false, false);
      acc[j] = __builtin_amdgcn_wmma_f32_16x16x32_bf16(false, ah, false, bl, (short)0, acc[j], false, false);
      acc[j] = __builtin_amdgcn_wmma_f32_16x16x32_bf16(false, al, false, bh, (short)0, acc[j], false, false);
    }
  }
  // ---- C tile store (ISA layout): VGPR r, lane L -> M=m0+r+8*half, N=n0+L%16
#pragma unroll
  for (int j = 0; j < NT; ++j) {
    const int cn = ((tg * NT + j) << 4) + mr;
#pragma unroll
    for (int r = 0; r < 8; ++r) {
      int cm = m0 + r + half * 8;
      if (cm < M) C[(long long)cm * ldC + cn] = acc[j][r];
    }
  }
}

// ---------------- layer 2 projection + attention scores (fused) ------------
// H2[n,0:2] = H[n,:] @ W2 ; aS[n] = H2[n,:]·as2 ; aD[n] = H2[n,:]·ad2
// Wave per node: lane l reads H[n, 16l .. 16l+15] (4x b128), W2 rows 16l..,
// wave32 shfl_xor tree reduction.
__global__ __launch_bounds__(256) void layer2_proj_kernel(
    const float* __restrict__ H, const float* __restrict__ W2,
    const float* __restrict__ as2, const float* __restrict__ ad2,
    float* __restrict__ H2, float* __restrict__ aS, float* __restrict__ aD,
    long long n_nodes)
{
  const int lane = threadIdx.x & 31;
  long long n = (long long)blockIdx.x * (blockDim.x >> 5) + (threadIdx.x >> 5);
  if (n >= n_nodes) return;
  const float* hp = H + n * 512 + lane * 16;
  const float* wp = W2 + (long long)lane * 16 * 2;   // rows 16l.., 2 cols each
  float d0 = 0.f, d1 = 0.f;
#pragma unroll
  for (int q = 0; q < 4; ++q) {
    float4 v   = *(const float4*)(hp + q * 4);
    float4 w01 = *(const float4*)(wp + q * 8);       // rows 4q, 4q+1
    float4 w23 = *(const float4*)(wp + q * 8 + 4);   // rows 4q+2, 4q+3
    d0 += v.x * w01.x + v.y * w01.z + v.z * w23.x + v.w * w23.z;
    d1 += v.x * w01.y + v.y * w01.w + v.z * w23.y + v.w * w23.w;
  }
#pragma unroll
  for (int off = 16; off > 0; off >>= 1) {
    d0 += __shfl_xor(d0, off, 32);
    d1 += __shfl_xor(d1, off, 32);
  }
  if (lane == 0) {
    H2[n * 2 + 0] = d0;
    H2[n * 2 + 1] = d1;
    aS[n] = d0 * as2[0] + d1 * as2[1];
    aD[n] = d0 * ad2[0] + d1 * ad2[1];
  }
}

// ---------------- per-node attention scores a_src/a_dst (layer 1) ----------
__global__ void att_scores_kernel(const float* __restrict__ H,
    const float* __restrict__ att_s, const float* __restrict__ att_d,
    float* __restrict__ as_out, float* __restrict__ ad_out,
    long long n_nodes, int heads, int C)
{
  long long i = (long long)blockIdx.x * blockDim.x + threadIdx.x;
  if (i >= n_nodes * heads) return;
  long long n = i / heads;
  int h = (int)(i % heads);
  const float* hp = H + n * heads * C + (long long)h * C;
  const float* sp = att_s + (long long)h * C;
  const float* dp = att_d + (long long)h * C;
  float ss = 0.f, sd = 0.f;
  for (int c = 0; c < C; ++c) { float v = hp[c]; ss += v * sp[c]; sd += v * dp[c]; }
  as_out[i] = ss;
  ad_out[i] = sd;
}

// ---------------- order-preserving float<->uint for atomicMax ----------------
__device__ __forceinline__ unsigned f2ord(float f) {
  unsigned u = __float_as_uint(f);
  return (u & 0x80000000u) ? ~u : (u | 0x80000000u);
}
__device__ __forceinline__ float ord2f(unsigned u) {
  u = (u & 0x80000000u) ? (u & 0x7fffffffu) : ~u;
  return __uint_as_float(u);
}
// encoded(-inf) = ~0xFF800000 = 0x007FFFFF (host-side constant)
#define ORD_NEG_INF 0x007FFFFFu

__device__ __forceinline__ void edge_endpoints(long long e, long long E,
    const long long* __restrict__ src, const long long* __restrict__ dst,
    long long& s, long long& d)
{
  if (e < E) { s = src[e]; d = dst[e]; } else { s = d = e - E; }
}

__device__ __forceinline__ float lrelu(float a) { return a > 0.f ? a : 0.2f * a; }

// ---------------- pass 1: segment max of attention logits ----------------
__global__ void edge_max_kernel(const long long* __restrict__ src,
    const long long* __restrict__ dst, long long E, long long n_nodes,
    const float* __restrict__ as_in, const float* __restrict__ ad_in,
    unsigned* __restrict__ m, int heads)
{
  long long i = (long long)blockIdx.x * blockDim.x + threadIdx.x;
  long long tot = (E + n_nodes) * heads;
  if (i >= tot) return;
  long long e = i / heads; int h = (int)(i % heads);
  long long s, d; edge_endpoints(e, E, src, dst, s, d);
  float a = lrelu(as_in[s * heads + h] + ad_in[d * heads + h]);
  atomicMax(&m[d * heads + h], f2ord(a));
}

// ---------------- pass 2: segment sum of exp(alpha - max) ----------------
__global__ void edge_expsum_kernel(const long long* __restrict__ src,
    const long long* __restrict__ dst, long long E, long long n_nodes,
    const float* __restrict__ as_in, const float* __restrict__ ad_in,
    const unsigned* __restrict__ m, float* __restrict__ denom, int heads)
{
  long long i = (long long)blockIdx.x * blockDim.x + threadIdx.x;
  long long tot = (E + n_nodes) * heads;
  if (i >= tot) return;
  long long e = i / heads; int h = (int)(i % heads);
  long long s, d; edge_endpoints(e, E, src, dst, s, d);
  float a = lrelu(as_in[s * heads + h] + ad_in[d * heads + h]);
  float ea = expf(a - ord2f(m[d * heads + h]));
  atomicAdd(&denom[d * heads + h], ea);
}

// ---------------- pass 3: weighted scatter of source features ----------------
__global__ void edge_agg_kernel(const long long* __restrict__ src,
    const long long* __restrict__ dst, long long E, long long n_nodes,
    const float* __restrict__ as_in, const float* __restrict__ ad_in,
    const unsigned* __restrict__ m, const float* __restrict__ denom,
    const float* __restrict__ H, float* __restrict__ acc, int heads, int C)
{
  long long i = (long long)blockIdx.x * blockDim.x + threadIdx.x;
  long long tot = (E + n_nodes) * heads;
  if (i >= tot) return;
  long long e = i / heads; int h = (int)(i % heads);
  long long s, d; edge_endpoints(e, E, src, dst, s, d);
  float a = lrelu(as_in[s * heads + h] + ad_in[d * heads + h]);
  float coef = expf(a - ord2f(m[d * heads + h])) / denom[d * heads + h];
  const float* hp = H + (s * heads + h) * (long long)C;
  float* op = acc + (d * heads + h) * (long long)C;
  if ((C & 3) == 0) {
    for (int c = 0; c < C; c += 4) {
      float4 v = *(const float4*)(hp + c);
      atomicAdd(op + c + 0, v.x * coef);
      atomicAdd(op + c + 1, v.y * coef);
      atomicAdd(op + c + 2, v.z * coef);
      atomicAdd(op + c + 3, v.w * coef);
    }
  } else {
    for (int c = 0; c < C; ++c) atomicAdd(op + c, hp[c] * coef);
  }
}

// ---------------- bias + ELU (layer 1 epilogue) ----------------
__global__ void bias_elu_kernel(const float* __restrict__ acc,
    const float* __restrict__ b, float* __restrict__ out, long long n, int F)
{
  long long i = (long long)blockIdx.x * blockDim.x + threadIdx.x;
  if (i >= n) return;
  float v = acc[i] + b[i % F];
  out[i] = v > 0.f ? v : expm1f(v);
}

// ---------------- bias add (final epilogue) ----------------
__global__ void add_bias_kernel(const float* __restrict__ acc,
    const float* __restrict__ b, float* __restrict__ out, long long n, int F)
{
  long long i = (long long)blockIdx.x * blockDim.x + threadIdx.x;
  if (i >= n) return;
  out[i] = acc[i] + b[i % F];
}

// ---------------------------------------------------------------------------
static inline int gridFor(long long n, int block) { return (int)((n + block - 1) / block); }

extern "C" void kernel_launch(void* const* d_in, const int* in_sizes, int n_in,
                              void* d_out, int out_size, void* d_ws, size_t ws_size,
                              hipStream_t stream) {
  const float*     x   = (const float*)d_in[0];
  const long long* ei  = (const long long*)d_in[1];   // int64 [2,E]
  const float*     W1  = (const float*)d_in[2];       // [128, 512]
  const float*     as1 = (const float*)d_in[3];       // [8, 64]
  const float*     ad1 = (const float*)d_in[4];       // [8, 64]
  const float*     b1  = (const float*)d_in[5];       // [512]
  const float*     W2  = (const float*)d_in[6];       // [512, 2]
  const float*     as2 = (const float*)d_in[7];       // [1, 2]
  const float*     ad2 = (const float*)d_in[8];       // [1, 2]
  const float*     b2  = (const float*)d_in[9];       // [2]

  const int F_IN = 128, HID = 64, HEADS = 8, NC = 2;
  const long long Nn = in_sizes[0] / F_IN;   // 20000 nodes
  const long long E  = in_sizes[1] / 2;      // 320000 edges
  const long long Et = E + Nn;               // + self loops
  const long long F1 = (long long)HEADS * HID; // 512

  const long long* src = ei;
  const long long* dst = ei + E;

  // ---- workspace carve-up (floats) ----
  float* ws = (float*)d_ws;
  float*    H1    = ws; ws += Nn * F1;          // [N,512] layer-1 features (then ELU'd in place)
  float*    ACC1  = ws; ws += Nn * F1;          // [N,512] aggregation accumulator
  float*    aS1   = ws; ws += Nn * HEADS;
  float*    aD1   = ws; ws += Nn * HEADS;
  unsigned* M1    = (unsigned*)ws; ws += Nn * HEADS;
  float*    DEN1  = ws; ws += Nn * HEADS;
  float*    H2    = ws; ws += Nn * NC;          // [N,2] layer-2 features
  float*    aS2   = ws; ws += Nn;
  float*    aD2   = ws; ws += Nn;
  unsigned* M2    = (unsigned*)ws; ws += Nn;
  float*    DEN2  = ws; ws += Nn;
  float*    ACC2  = ws; ws += Nn * NC;

  const int BLK = 256;

  // ================= Layer 1 =================
  // GEMM1: H1 = X @ W1   (20000x128 @ 128x512) — WMMA bf16x3, 4 N-tiles/wave
  {
    long long tilesM = (Nn + 15) / 16, nGroups = (F1 / 16) / 4;
    long long waves = tilesM * nGroups;
    gemm_wmma_bf16x3<4><<<gridFor(waves * 32, BLK), BLK, 0, stream>>>(
        x, W1, H1, (int)Nn, (int)F1, F_IN, F_IN, (int)F1, (int)F1);
  }
  att_scores_kernel<<<gridFor(Nn * HEADS, BLK), BLK, 0, stream>>>(
      H1, as1, ad1, aS1, aD1, Nn, HEADS, HID);

  fill_u32_kernel<<<gridFor(Nn * HEADS, BLK), BLK, 0, stream>>>(M1, ORD_NEG_INF, Nn * HEADS);
  fill_u32_kernel<<<gridFor(Nn * HEADS, BLK), BLK, 0, stream>>>((unsigned*)DEN1, 0u, Nn * HEADS);
  fill_u32_kernel<<<gridFor(Nn * F1, BLK), BLK, 0, stream>>>((unsigned*)ACC1, 0u, Nn * F1);

  edge_max_kernel<<<gridFor(Et * HEADS, BLK), BLK, 0, stream>>>(
      src, dst, E, Nn, aS1, aD1, M1, HEADS);
  edge_expsum_kernel<<<gridFor(Et * HEADS, BLK), BLK, 0, stream>>>(
      src, dst, E, Nn, aS1, aD1, M1, DEN1, HEADS);
  edge_agg_kernel<<<gridFor(Et * HEADS, BLK), BLK, 0, stream>>>(
      src, dst, E, Nn, aS1, aD1, M1, DEN1, H1, ACC1, HEADS, HID);

  // H1 <- elu(ACC1 + b1)   (layer-1 output, input to layer 2)
  bias_elu_kernel<<<gridFor(Nn * F1, BLK), BLK, 0, stream>>>(ACC1, b1, H1, Nn * F1, (int)F1);

  // ================= Layer 2 =================
  // Fused projection + attention scores (memory-bound; wave-per-node)
  layer2_proj_kernel<<<gridFor(Nn * 32, BLK), BLK, 0, stream>>>(
      H1, W2, as2, ad2, H2, aS2, aD2, Nn);

  fill_u32_kernel<<<gridFor(Nn, BLK), BLK, 0, stream>>>(M2, ORD_NEG_INF, Nn);
  fill_u32_kernel<<<gridFor(Nn, BLK), BLK, 0, stream>>>((unsigned*)DEN2, 0u, Nn);
  fill_u32_kernel<<<gridFor(Nn * NC, BLK), BLK, 0, stream>>>((unsigned*)ACC2, 0u, Nn * NC);

  edge_max_kernel<<<gridFor(Et, BLK), BLK, 0, stream>>>(
      src, dst, E, Nn, aS2, aD2, M2, 1);
  edge_expsum_kernel<<<gridFor(Et, BLK), BLK, 0, stream>>>(
      src, dst, E, Nn, aS2, aD2, M2, DEN2, 1);
  edge_agg_kernel<<<gridFor(Et, BLK), BLK, 0, stream>>>(
      src, dst, E, Nn, aS2, aD2, M2, DEN2, H2, ACC2, 1, NC);

  // out = ACC2 (mean over 1 head == identity) + b2
  add_bias_kernel<<<gridFor(Nn * NC, BLK), BLK, 0, stream>>>(
      ACC2, b2, (float*)d_out, Nn * NC, NC);
}
